// LinearModel2_81003083203432
// MI455X (gfx1250) — matrix operand order repaired and checked by reading, not compiled, not run on hardware
//
#include <hip/hip_runtime.h>
#include <hip/hip_bf16.h>

// MI455X / gfx1250, wave32. WMMA f16 16x16x32 (f32 acc) + TDM tensor_load_to_lds.

typedef __attribute__((ext_vector_type(16))) _Float16 v16h;
typedef __attribute__((ext_vector_type(8)))  float    v8f;
typedef __attribute__((ext_vector_type(4)))  unsigned int v4u;
typedef __attribute__((ext_vector_type(8)))  int      v8i;
typedef __attribute__((ext_vector_type(4)))  int      v4i;

#define NROW 4096
#define NDIMM 1024
#define DIN  64
#define NOUT 16
#define NS_ITERS 10
#define SIGMA 0.2857142857f  // 2/(lambda_min+lambda_max) with spectrum ~[1,6]

// ---- WMMA fragment loaders (CDNA5 ISA 7.12.2; 16-bit A 16x32, B 32x16) ----

__device__ __forceinline__ v16h load_a_f16(const _Float16* tile, int ld, int lane) {
  const int h  = lane >> 4;
  const int m  = lane & 15;
  const int kb = h * 8;
  const _Float16* p = tile + (size_t)m * ld;
  v16h a;
#pragma unroll
  for (int j = 0; j < 8; ++j) {
    a[j]     = p[kb + j];
    a[8 + j] = p[16 + kb + j];
  }
  return a;
}

__device__ __forceinline__ v16h load_b_f16(const _Float16* tile, int ld, int lane) {
  const int h  = lane >> 4;
  const int n  = lane & 15;
  const int kb = h * 8;
  v16h b;
#pragma unroll
  for (int j = 0; j < 8; ++j) {
    b[j]     = tile[(size_t)(kb + j) * ld + n];
    b[8 + j] = tile[(size_t)(16 + kb + j) * ld + n];
  }
  return b;
}

// ---- TDM: async 2D f16 tile load Global -> LDS (ISA ch.8 D# layout) ----
// tile_cols = contiguous width (elements), tile_rows = #rows,
// src_ld = tensor row stride (elements), tensor_rows = rows in tensor (OOB bound).
__device__ __forceinline__ void tdm_load_2d_f16(const _Float16* gsrc, unsigned lds_byte_off,
                                                int tile_cols, int tile_rows,
                                                int src_ld, int tensor_rows) {
  const unsigned long long ga = (unsigned long long)(uintptr_t)gsrc;
  v4u g0;
  g0[0] = 1u;                                        // count=1, user descriptor
  g0[1] = lds_byte_off;                              // LDS byte address
  g0[2] = (unsigned)(ga & 0xffffffffu);              // global_addr[31:0]
  g0[3] = (unsigned)((ga >> 32) & 0x1ffffffu) | (2u << 30);  // addr[56:32] | type=2
  v8i g1;
  g1[0] = (1 << 16);                                 // wg_mask=0 | data_size=1 (2B)
  g1[1] = (src_ld & 0xffff) << 16;                   // tensor_dim0[15:0]
  g1[2] = ((src_ld >> 16) & 0xffff) | ((tensor_rows & 0xffff) << 16);  // dim0 hi | dim1 lo
  g1[3] = ((tensor_rows >> 16) & 0xffff) | (tile_cols << 16);          // dim1 hi | tile_dim0
  g1[4] = (tile_rows & 0xffff);                      // tile_dim1 | tile_dim2=0
  g1[5] = src_ld;                                    // tensor_dim0_stride[31:0]
  g1[6] = 0;                                         // stride hi | dim1_stride lo
  g1[7] = 0;
  const v4i gz = {0, 0, 0, 0};
#if __clang_major__ >= 23
  const v8i gz8 = {0, 0, 0, 0, 0, 0, 0, 0};
  __builtin_amdgcn_tensor_load_to_lds(g0, g1, gz, gz, gz8, 0);
#else
  __builtin_amdgcn_tensor_load_to_lds(g0, g1, gz, gz, 0);
#endif
}

// ---- Kernel 1: F = tanh(x @ Wf), store f32 + f16 copies ----
__global__ void k_feature(const float* __restrict__ x, const float* __restrict__ Wf,
                          float* __restrict__ F, _Float16* __restrict__ Fh) {
  const int lane = threadIdx.x & 31;
  const int wave = threadIdx.x >> 5;
  const int row0 = blockIdx.x * 16;
  const int col0 = (blockIdx.y * 4 + wave) * 16;
  v8f c = {};
#pragma unroll
  for (int k0 = 0; k0 < DIN; k0 += 32) {
    v16h a, b;
    {
      const int h = lane >> 4, m = lane & 15, kb = h * 8;
      const float* p = x + (size_t)(row0 + m) * DIN + k0;
#pragma unroll
      for (int j = 0; j < 8; ++j) {
        a[j]     = (_Float16)p[kb + j];
        a[8 + j] = (_Float16)p[16 + kb + j];
      }
    }
    {
      const int h = lane >> 4, n = lane & 15, kb = h * 8;
      const float* p = Wf + (size_t)k0 * NDIMM + col0 + n;
#pragma unroll
      for (int j = 0; j < 8; ++j) {
        b[j]     = (_Float16)p[(size_t)(kb + j) * NDIMM];
        b[8 + j] = (_Float16)p[(size_t)(16 + kb + j) * NDIMM];
      }
    }
    c = __builtin_amdgcn_wmma_f32_16x16x32_f16(false, a, false, b, (short)0, c,
                                               false, false);
  }
#pragma unroll
  for (int r = 0; r < 8; ++r) {
    const int m = r + ((lane >> 4) << 3);
    const int n = lane & 15;
    const float v = tanhf(c[r]);
    const size_t idx = (size_t)(row0 + m) * NDIMM + col0 + n;
    F[idx]  = v;
    Fh[idx] = (_Float16)v;
  }
}

// ---- Kernel 2: m[n,k] = sum_i F[n,i] * w[k,i] ----
__global__ void k_m(const float* __restrict__ F, const float* __restrict__ w,
                    float* __restrict__ out_m) {
  const int n = blockIdx.x * blockDim.x + threadIdx.x;
  float acc[NOUT];
#pragma unroll
  for (int k = 0; k < NOUT; ++k) acc[k] = 0.0f;
  for (int i = 0; i < NDIMM; ++i) {
    const float f = F[(size_t)n * NDIMM + i];
#pragma unroll
    for (int k = 0; k < NOUT; ++k) acc[k] += f * w[(size_t)k * NDIMM + i];
  }
#pragma unroll
  for (int k = 0; k < NOUT; ++k) out_m[(size_t)n * NOUT + k] = acc[k];
}

// ---- Kernel 3: Ph = f16(P);  X = sigma*I (f32 + f16) ----
__global__ void k_init(const float* __restrict__ P, _Float16* __restrict__ Ph,
                       float* __restrict__ X, _Float16* __restrict__ Xh) {
  const size_t idx = (size_t)blockIdx.x * blockDim.x + threadIdx.x;
  Ph[idx] = (_Float16)P[idx];
  const size_t in_mat = idx & (size_t)(NDIMM * NDIMM - 1);
  const int i = (int)(in_mat >> 10);
  const int j = (int)(in_mat & (NDIMM - 1));
  const float x0 = (i == j) ? SIGMA : 0.0f;
  X[idx]  = x0;
  Xh[idx] = (_Float16)x0;
}

// ---- Kernel 4: batched C = A @ B, TDM double-buffered LDS panels ----
// Block 256 (8 waves) computes a 128x64 tile: wave w owns M-slot w, 4 N-tiles.
// grid: (NDIMM/128, NDIMM/64, NOUT).
#define BM 128
#define BN 64
#define BK 32
__global__ void k_gemm(const _Float16* __restrict__ A, const _Float16* __restrict__ B,
                       float* __restrict__ Cf, _Float16* __restrict__ Ch) {
  __shared__ __align__(16) _Float16 sA[2][BM * BK];  // 2 x 8 KB
  __shared__ __align__(16) _Float16 sB[2][BK * BN];  // 2 x 4 KB
  const int lane = threadIdx.x & 31;
  const int wave = threadIdx.x >> 5;
  const size_t bo = (size_t)blockIdx.z * NDIMM * NDIMM;
  const int row0 = blockIdx.x * BM;
  const int col0 = blockIdx.y * BN;
  const _Float16* Ab = A + bo + (size_t)row0 * NDIMM;
  const _Float16* Bb = B + bo + col0;
  const int NK = NDIMM / BK;

  v8f acc[4];
#pragma unroll
  for (int nt = 0; nt < 4; ++nt) acc[nt] = (v8f){};

  if (wave == 0) {  // stage 0 panels via Tensor Data Mover
    tdm_load_2d_f16(Ab, (unsigned)(uintptr_t)&sA[0][0], BK, BM, NDIMM, NDIMM);
    tdm_load_2d_f16(Bb, (unsigned)(uintptr_t)&sB[0][0], BN, BK, NDIMM, NDIMM);
    __builtin_amdgcn_s_wait_tensorcnt((short)0);
  }
  __syncthreads();

  for (int kt = 0; kt < NK; ++kt) {
    const int cur = kt & 1;
    if (wave == 0 && kt + 1 < NK) {  // prefetch next panels into the other buffer
      tdm_load_2d_f16(Ab + (kt + 1) * BK, (unsigned)(uintptr_t)&sA[1 - cur][0],
                      BK, BM, NDIMM, NDIMM);
      tdm_load_2d_f16(Bb + (size_t)(kt + 1) * BK * NDIMM,
                      (unsigned)(uintptr_t)&sB[1 - cur][0], BN, BK, NDIMM, NDIMM);
    }
    const v16h a = load_a_f16(&sA[cur][wave * 16 * BK], BK, lane);
#pragma unroll
    for (int nt = 0; nt < 4; ++nt) {
      const v16h b = load_b_f16(&sB[cur][nt * 16], BN, lane);
      acc[nt] = __builtin_amdgcn_wmma_f32_16x16x32_f16(false, a, false, b, (short)0,
                                                       acc[nt], false, false);
    }
    if (wave == 0 && kt + 1 < NK) __builtin_amdgcn_s_wait_tensorcnt((short)0);
    __syncthreads();
  }

#pragma unroll
  for (int nt = 0; nt < 4; ++nt) {
#pragma unroll
    for (int r = 0; r < 8; ++r) {
      const int m = r + ((lane >> 4) << 3);
      const int n = lane & 15;
      const size_t idx = bo + (size_t)(row0 + wave * 16 + m) * NDIMM + col0 + nt * 16 + n;
      if (Cf) Cf[idx] = acc[nt][r];
      if (Ch) Ch[idx] = (_Float16)acc[nt][r];
    }
  }
}

// ---- Kernel 5: Newton-Schulz update  X = 2X - T ; Xh = f16(X) ----
__global__ void k_update(float* __restrict__ X, const float* __restrict__ T,
                         _Float16* __restrict__ Xh) {
  const size_t idx = (size_t)blockIdx.x * blockDim.x + threadIdx.x;
  const float v = 2.0f * X[idx] - T[idx];
  X[idx]  = v;
  Xh[idx] = (_Float16)v;
}

// ---- Kernel 6: fused s[n,k] = F[n,:] @ Pinv_k @ F[n,:]^T (diag only) ----
__global__ void k_s(const _Float16* __restrict__ Fh, const _Float16* __restrict__ Pinvh,
                    float* __restrict__ out_s) {
  __shared__ __align__(16) _Float16 sF[16 * NDIMM];  // 32 KB
  __shared__ float s_acc[16];
  const int k    = blockIdx.y;
  const int row0 = blockIdx.x * 16;
  {
    const uint4* src = (const uint4*)(Fh + (size_t)row0 * NDIMM);
    uint4* dst = (uint4*)sF;
    for (int i = threadIdx.x; i < (16 * NDIMM) / 8; i += blockDim.x) dst[i] = src[i];
  }
  if (threadIdx.x < 16) s_acc[threadIdx.x] = 0.0f;
  __syncthreads();

  const int lane = threadIdx.x & 31;
  const int wave = threadIdx.x >> 5;
  const _Float16* Pk = Pinvh + (size_t)k * NDIMM * NDIMM;

  float part[8];
#pragma unroll
  for (int r = 0; r < 8; ++r) part[r] = 0.0f;

  for (int t = 0; t < NDIMM / (16 * 8); ++t) {
    const int col0 = (t * 8 + wave) * 16;
    v8f c = {};
    for (int k0 = 0; k0 < NDIMM; k0 += 32) {
      v16h a = load_a_f16(sF + k0, NDIMM, lane);                        // LDS
      v16h b = load_b_f16(Pk + (size_t)k0 * NDIMM + col0, NDIMM, lane); // L2
      c = __builtin_amdgcn_wmma_f32_16x16x32_f16(false, a, false, b, (short)0, c,
                                                 false, false);
    }
#pragma unroll
    for (int r = 0; r < 8; ++r) {
      const int m = r + ((lane >> 4) << 3);
      const int n = lane & 15;
      part[r] += c[r] * (float)sF[(size_t)m * NDIMM + col0 + n];
    }
  }
#pragma unroll
  for (int off = 1; off < 16; off <<= 1) {
#pragma unroll
    for (int r = 0; r < 8; ++r) part[r] += __shfl_xor(part[r], off, 32);
  }
  if ((lane & 15) == 0) {
    const int mbase = (lane >> 4) * 8;
#pragma unroll
    for (int r = 0; r < 8; ++r) atomicAdd(&s_acc[mbase + r], part[r]);
  }
  __syncthreads();
  if (threadIdx.x < 16)
    out_s[(size_t)(row0 + threadIdx.x) * NOUT + k] = s_acc[threadIdx.x];
}

// ---- Host orchestration ----
extern "C" void kernel_launch(void* const* d_in, const int* in_sizes, int n_in,
                              void* d_out, int out_size, void* d_ws, size_t ws_size,
                              hipStream_t stream) {
  (void)in_sizes; (void)n_in; (void)out_size; (void)ws_size;
  const float* x  = (const float*)d_in[0];   // (4096, 64)
  const float* Wf = (const float*)d_in[1];   // (64, 1024)
  const float* w  = (const float*)d_in[2];   // (16, 1024, 1)
  const float* P  = (const float*)d_in[3];   // (16, 1024, 1024)

  float* out_m = (float*)d_out;                       // (4096, 16)
  float* out_s = out_m + (size_t)NROW * NOUT;         // (4096, 16)

  char* ws = (char*)d_ws;
  float*    F  = (float*)ws;     ws += (size_t)NROW * NDIMM * sizeof(float);
  _Float16* Fh = (_Float16*)ws;  ws += (size_t)NROW * NDIMM * sizeof(_Float16);
  _Float16* Ph = (_Float16*)ws;  ws += (size_t)NOUT * NDIMM * NDIMM * sizeof(_Float16);
  float*    X  = (float*)ws;     ws += (size_t)NOUT * NDIMM * NDIMM * sizeof(float);
  _Float16* Xh = (_Float16*)ws;  ws += (size_t)NOUT * NDIMM * NDIMM * sizeof(_Float16);
  _Float16* Yh = (_Float16*)ws;  ws += (size_t)NOUT * NDIMM * NDIMM * sizeof(_Float16);
  float*    T  = (float*)ws;     ws += (size_t)NOUT * NDIMM * NDIMM * sizeof(float);

  const size_t tot = (size_t)NOUT * NDIMM * NDIMM;

  k_init<<<dim3((unsigned)(tot / 256)), dim3(256), 0, stream>>>(P, Ph, X, Xh);
  k_feature<<<dim3(NROW / 16, NDIMM / 64), dim3(128), 0, stream>>>(x, Wf, F, Fh);
  k_m<<<dim3(NROW / 256), dim3(256), 0, stream>>>(F, w, out_m);

  for (int it = 0; it < NS_ITERS; ++it) {
    // Y = P @ X  (f16 out only)
    k_gemm<<<dim3(NDIMM / BM, NDIMM / BN, NOUT), dim3(256), 0, stream>>>(Ph, Xh, nullptr, Yh);
    // T = X @ Y  (f32 out)
    k_gemm<<<dim3(NDIMM / BM, NDIMM / BN, NOUT), dim3(256), 0, stream>>>(Xh, Yh, T, nullptr);
    // X = 2X - T (f32 master), refresh f16 copy
    k_update<<<dim3((unsigned)(tot / 256)), dim3(256), 0, stream>>>(X, T, Xh);
  }

  k_s<<<dim3(NROW / 16, NOUT), dim3(256), 0, stream>>>(Fh, Xh, out_s);
}
